// RelationalGraphConvolutionalNetwork_m_6055903887371
// MI455X (gfx1250) — matrix operand order repaired and checked by reading, not compiled, not run on hardware
//
#include <hip/hip_runtime.h>
#include <hip/hip_bf16.h>
#include <stdint.h>
#include <stddef.h>

// ---------------- problem constants ----------------
#define NUM_NODE   32768
#define NUM_RES    4096
#define NUM_GRAPH  8
#define NUM_EDGE   262144
#define NUM_REL    7
#define D          256
#define KREL       (NUM_REL * D)   // 1792
#define KTOT       (KREL + D)      // 2048
#define PAD_ROWS   1024
#define N_SITES    4
#define EPS_F      1e-10f

typedef _Float16 v8h  __attribute__((ext_vector_type(8)));
typedef _Float16 v16h __attribute__((ext_vector_type(16)));
typedef float    v8f  __attribute__((ext_vector_type(8)));

// ---------------- utility: zero fill ----------------
__global__ void zero_f32(float* __restrict__ p, size_t n) {
    size_t i = (size_t)blockIdx.x * blockDim.x + threadIdx.x;
    size_t stride = (size_t)gridDim.x * blockDim.x;
    for (; i < n; i += stride) p[i] = 0.0f;
}

// ---------------- edge scatter: num/den segment sums ----------------
// one wave (32 lanes) per edge; lane covers 8 consecutive feature dwords
__global__ void __launch_bounds__(256) edge_scatter(
    const float* __restrict__ h,
    const int*   __restrict__ ein,
    const int*   __restrict__ eout,
    const int*   __restrict__ erel,
    const float* __restrict__ ew,
    float* __restrict__ num,
    float* __restrict__ den)
{
    const int wave = threadIdx.x >> 5;
    const int lane = threadIdx.x & 31;
    const int e = blockIdx.x * 8 + wave;
    const int src = ein[e];
    const int seg = eout[e] * NUM_REL + erel[e];
    const float w = ew[e];

    const float4* hp = (const float4*)(h + (size_t)src * D + lane * 8);
    float4 a = hp[0];
    float4 b = hp[1];
    float* np = num + (size_t)seg * D + lane * 8;
    atomicAdd(np + 0, a.x * w);
    atomicAdd(np + 1, a.y * w);
    atomicAdd(np + 2, a.z * w);
    atomicAdd(np + 3, a.w * w);
    atomicAdd(np + 4, b.x * w);
    atomicAdd(np + 5, b.y * w);
    atomicAdd(np + 6, b.z * w);
    atomicAdd(np + 7, b.w * w);
    if (lane == 0) atomicAdd(den + seg, w);
}

// ---------------- fuse /(den+eps) + f32->f16 into combined A = [update | h] ----------------
__global__ void __launch_bounds__(256) prep_A(
    const float* __restrict__ num,
    const float* __restrict__ den,
    const float* __restrict__ h,
    _Float16* __restrict__ A)
{
    const int n  = blockIdx.x;
    const int k0 = threadIdx.x * 8;
    v8h out;
    if (k0 < KREL) {
        const int r = k0 >> 8;                 // 8 | 256, so r constant across chunk
        const float inv = 1.0f / (den[(size_t)n * NUM_REL + r] + EPS_F);
        const float* src = num + (size_t)n * KREL + k0;
        #pragma unroll
        for (int i = 0; i < 8; ++i) out[i] = (_Float16)(src[i] * inv);
    } else {
        const float* src = h + (size_t)n * D + (k0 - KREL);
        #pragma unroll
        for (int i = 0; i < 8; ++i) out[i] = (_Float16)src[i];
    }
    *(v8h*)(A + (size_t)n * KTOT + k0) = out;
}

// ---------------- pack W = [W_rel | W_loop] into f16, fold biases ----------------
__global__ void __launch_bounds__(256) prep_W(
    const float* __restrict__ W_rel,
    const float* __restrict__ W_loop,
    const float* __restrict__ b0,
    const float* __restrict__ b1,
    _Float16* __restrict__ Wc,
    float* __restrict__ bias)
{
    const int j  = blockIdx.x;
    const int k0 = threadIdx.x * 8;
    v8h out;
    if (k0 < KREL) {
        const float* src = W_rel + (size_t)j * KREL + k0;
        #pragma unroll
        for (int i = 0; i < 8; ++i) out[i] = (_Float16)src[i];
    } else {
        const float* src = W_loop + (size_t)j * D + (k0 - KREL);
        #pragma unroll
        for (int i = 0; i < 8; ++i) out[i] = (_Float16)src[i];
    }
    *(v8h*)(Wc + (size_t)j * KTOT + k0) = out;
    if (threadIdx.x == 0) bias[j] = b0[j] + b1[j];
}

// ---------------- WMMA GEMM: out = relu(A(f16) x Wc^T(f16) + bias), f32 accumulate ----------------
// block = 256 threads = 8 waves; block tile 64(M) x 256(N); wave tile 32(M) x 64(N)
// LDS staging uses CDNA5 async global->LDS copies (ASYNCcnt), no VGPR round trip.
#define BM   64
#define LDA  40   // padded LDS row stride in halves (80B, spreads banks)
#define LDB  40
__global__ void __launch_bounds__(256) rgcn_gemm(
    const _Float16* __restrict__ A,
    const _Float16* __restrict__ W,
    const float* __restrict__ bias,
    float* __restrict__ out)
{
    __shared__ _Float16 As[BM * LDA];   //  5,120 B
    __shared__ _Float16 Bs[D * LDB];    // 20,480 B

    const int tid    = threadIdx.x;
    const int wave   = tid >> 5;
    const int lane   = tid & 31;
    const int wm     = wave >> 2;   // 0..1 (M strip of 32)
    const int wn     = wave & 3;    // 0..3 (N strip of 64)
    const int lane16 = lane & 15;
    const int g      = lane >> 4;   // hi/lo lane group
    const int row0   = blockIdx.x * BM;

    // this thread's staging slice: 16B of A, 64B (one row) of B per k-step
    const int ar = tid >> 2;
    const int ac = (tid & 3) * 8;
    const _Float16* gA = A + (size_t)(row0 + ar) * KTOT + ac;
    const _Float16* gB = W + (size_t)tid * KTOT;
    const unsigned ldsA = (unsigned)(size_t)&As[ar * LDA + ac];
    const unsigned ldsB = (unsigned)(size_t)&Bs[tid * LDB];

    v8f acc[2][4];
    #pragma unroll
    for (int mt = 0; mt < 2; ++mt)
        #pragma unroll
        for (int nt = 0; nt < 4; ++nt)
            #pragma unroll
            for (int v = 0; v < 8; ++v) acc[mt][nt][v] = 0.0f;

    for (int ks = 0; ks < KTOT; ks += 32) {
        __syncthreads();    // prior-iteration LDS reads complete across block

        // CDNA5 async copy global -> LDS (tracked by ASYNCcnt).
        // INST_OFFSET is added to BOTH the LDS and global addresses, so the
        // four contiguous 16B chunks of each B row share one base pair.
        asm volatile("global_load_async_to_lds_b128 %0, %1, off"
                     :: "v"(ldsA), "v"(gA) : "memory");
        asm volatile("global_load_async_to_lds_b128 %0, %1, off\n\t"
                     "global_load_async_to_lds_b128 %0, %1, off offset:16\n\t"
                     "global_load_async_to_lds_b128 %0, %1, off offset:32\n\t"
                     "global_load_async_to_lds_b128 %0, %1, off offset:48"
                     :: "v"(ldsB), "v"(gB) : "memory");
        if (ks + 64 < KTOT) {   // prefetch the k-step after next into GL2/L0
            __builtin_prefetch((const void*)(gA + 64), 0, 1);
            __builtin_prefetch((const void*)(gB + 64), 0, 1);
        }
        gA += 32;
        gB += 32;

        asm volatile("s_wait_asynccnt 0x0" ::: "memory");
        __syncthreads();    // all waves' async transfers now visible

        // A fragments per ISA 16-bit A 16x32 layout:
        // e<8 : k = g*8 + e ; e>=8 : k = 16 + g*8 + (e-8)
        v16h afrag[2];
        #pragma unroll
        for (int mt = 0; mt < 2; ++mt) {
            const _Float16* ab = &As[(wm * 32 + mt * 16 + lane16) * LDA];
            v8h lo = *(const v8h*)(ab + g * 8);
            v8h hi = *(const v8h*)(ab + 16 + g * 8);
            #pragma unroll
            for (int i = 0; i < 8; ++i) { afrag[mt][i] = lo[i]; afrag[mt][8 + i] = hi[i]; }
        }

        #pragma unroll
        for (int nt = 0; nt < 4; ++nt) {
            // B fragment: e<8 : k = g*8 + e ; e>=8 : k = 16 + g*8 + (e-8)
            const int col = wn * 64 + nt * 16 + lane16;
            const _Float16* bb = &Bs[col * LDB];
            v8h blo = *(const v8h*)(bb + g * 8);
            v8h bhi = *(const v8h*)(bb + 16 + g * 8);
            v16h bfrag;
            #pragma unroll
            for (int i = 0; i < 8; ++i) { bfrag[i] = blo[i]; bfrag[8 + i] = bhi[i]; }
            #pragma unroll
            for (int mt = 0; mt < 2; ++mt) {
                acc[mt][nt] = __builtin_amdgcn_wmma_f32_16x16x32_f16(
                    false, afrag[mt], false, bfrag, (short)0, acc[mt][nt], false, false);
            }
        }
    }

    // epilogue: bias + relu; C/D layout: lane n = lane16, VGPR v -> m = v + 8*g
    #pragma unroll
    for (int mt = 0; mt < 2; ++mt) {
        const int rbase = row0 + wm * 32 + mt * 16 + g * 8;
        #pragma unroll
        for (int nt = 0; nt < 4; ++nt) {
            const int col = wn * 64 + nt * 16 + lane16;
            const float b = bias[col];
            #pragma unroll
            for (int v = 0; v < 8; ++v) {
                float val = acc[mt][nt][v] + b;
                out[(size_t)(rbase + v) * D + col] = fmaxf(val, 0.0f);
            }
        }
    }
}

// ---------------- residue pooling: atomic segment sums ----------------
// one wave per atom
__global__ void __launch_bounds__(256) atom_scatter(
    const float* __restrict__ h,
    const int*   __restrict__ a2r,
    float* __restrict__ rsum,
    float* __restrict__ rcnt)
{
    const int wave = threadIdx.x >> 5;
    const int lane = threadIdx.x & 31;
    const int a = blockIdx.x * 8 + wave;
    const int res = a2r[a];
    const float4* hp = (const float4*)(h + (size_t)a * D + lane * 8);
    float4 u = hp[0];
    float4 v = hp[1];
    float* rp = rsum + (size_t)res * D + lane * 8;
    atomicAdd(rp + 0, u.x); atomicAdd(rp + 1, u.y);
    atomicAdd(rp + 2, u.z); atomicAdd(rp + 3, u.w);
    atomicAdd(rp + 4, v.x); atomicAdd(rp + 5, v.y);
    atomicAdd(rp + 6, v.z); atomicAdd(rp + 7, v.w);
    if (lane == 0) atomicAdd(rcnt + res, 1.0f);
}

// ---------------- per-graph start offsets (counts -> exclusive prefix) ----------------
__global__ void graph_starts(const int* __restrict__ r2g, int* __restrict__ starts)
{
    __shared__ int cnt[NUM_GRAPH];
    if (threadIdx.x < NUM_GRAPH) cnt[threadIdx.x] = 0;
    __syncthreads();
    for (int r = threadIdx.x; r < NUM_RES; r += blockDim.x)
        atomicAdd(&cnt[r2g[r]], 1);
    __syncthreads();
    if (threadIdx.x == 0) {
        int acc = 0;
        for (int gg = 0; gg < NUM_GRAPH; ++gg) { starts[gg] = acc; acc += cnt[gg]; }
    }
}

// ---------------- write residue means into padded output ----------------
__global__ void __launch_bounds__(256) write_padded(
    const float* __restrict__ rsum,
    const float* __restrict__ rcnt,
    const int*   __restrict__ r2g,
    const int*   __restrict__ starts,
    float* __restrict__ padded)
{
    const int r = blockIdx.x;
    const int k = threadIdx.x;
    const int gg = r2g[r];
    const int pos = r - starts[gg];
    padded[((size_t)gg * PAD_ROWS + pos) * D + k] = rsum[(size_t)r * D + k] / rcnt[r];
}

// ---------------- gather site features ----------------
__global__ void __launch_bounds__(256) site_gather(
    const float* __restrict__ padded,
    const int*   __restrict__ positions,
    float* __restrict__ gf)
{
    const int idx = blockIdx.x;           // g*N_SITES + s
    const int gg = idx / N_SITES;
    const int p = positions[idx];
    gf[(size_t)idx * D + threadIdx.x] =
        padded[((size_t)gg * PAD_ROWS + p) * D + threadIdx.x];
}

// ---------------- host launcher ----------------
extern "C" void kernel_launch(void* const* d_in, const int* in_sizes, int n_in,
                              void* d_out, int out_size, void* d_ws, size_t ws_size,
                              hipStream_t stream)
{
    // Classify inputs by element count (robust to dict-insertion vs pytree-sorted order;
    // the four 262144-element edge arrays keep relative order under both; per-layer
    // bias pair is summed so its internal order is irrelevant).
    const float* x = nullptr;
    const float* W_rel[3]  = {nullptr, nullptr, nullptr};
    const float* W_loop[3] = {nullptr, nullptr, nullptr};
    const float* bvec[6]   = {nullptr, nullptr, nullptr, nullptr, nullptr, nullptr};
    const float* edge_weight = nullptr;
    const int *node_in = nullptr, *node_out = nullptr, *relation = nullptr;
    const int *atom2residue = nullptr, *residue2graph = nullptr, *positions = nullptr;
    int nwr = 0, nwl = 0, nb = 0, n262 = 0;
    for (int i = 0; i < n_in; ++i) {
        const int s = in_sizes[i];
        void* p = d_in[i];
        if      (s == NUM_NODE * D)  x = (const float*)p;
        else if (s == D * KREL)      { if (nwr < 3) W_rel[nwr++]  = (const float*)p; }
        else if (s == D * D)         { if (nwl < 3) W_loop[nwl++] = (const float*)p; }
        else if (s == D)             { if (nb  < 6) bvec[nb++]    = (const float*)p; }
        else if (s == NUM_EDGE) {
            if      (n262 == 0) edge_weight = (const float*)p;
            else if (n262 == 1) node_in  = (const int*)p;
            else if (n262 == 2) node_out = (const int*)p;
            else if (n262 == 3) relation = (const int*)p;
            ++n262;
        }
        else if (s == NUM_NODE)          atom2residue  = (const int*)p;
        else if (s == NUM_RES)           residue2graph = (const int*)p;
        else if (s == NUM_GRAPH*N_SITES) positions     = (const int*)p;
    }

    // workspace carve-up
    char* ws = (char*)d_ws;
    size_t off = 0;
    auto carve = [&](size_t bytes) -> void* {
        void* p = ws + off;
        off += (bytes + 255) & ~(size_t)255;
        return p;
    };
    float*    num  = (float*)   carve((size_t)NUM_NODE * KREL * 4);   // 224 MiB
    float*    den  = (float*)   carve((size_t)NUM_NODE * NUM_REL * 4);
    _Float16* Af   = (_Float16*)carve((size_t)NUM_NODE * KTOT * 2);   // 128 MiB
    _Float16* Wc   = (_Float16*)carve((size_t)D * KTOT * 2);
    float*    bias = (float*)   carve((size_t)D * 4);
    float*    h0   = (float*)   carve((size_t)NUM_NODE * D * 4);
    float*    h1   = (float*)   carve((size_t)NUM_NODE * D * 4);
    // pooling scratch overlays num (num is dead by pooling time)
    float*    rsum   = num;
    float*    rcnt   = num + (size_t)NUM_RES * D;
    int*      starts = (int*)(num + (size_t)NUM_RES * D + NUM_RES);

    float* out_f      = (float*)d_out;
    float* out_gf     = out_f;                              // 8*4*256
    float* out_padded = out_f + (size_t)NUM_GRAPH * N_SITES * D;

    const float* hin = x;
    float* hbuf[2] = {h0, h1};

    for (int L = 0; L < 3; ++L) {
        zero_f32<<<2048, 256, 0, stream>>>(num, (size_t)NUM_NODE * KREL);
        zero_f32<<<64,   256, 0, stream>>>(den, (size_t)NUM_NODE * NUM_REL);
        edge_scatter<<<NUM_EDGE / 8, 256, 0, stream>>>(
            hin, node_in, node_out, relation, edge_weight, num, den);
        prep_A<<<NUM_NODE, 256, 0, stream>>>(num, den, hin, Af);
        prep_W<<<D, 256, 0, stream>>>(W_rel[L], W_loop[L],
                                      bvec[2 * L], bvec[2 * L + 1], Wc, bias);
        float* hout = hbuf[L & 1];
        rgcn_gemm<<<NUM_NODE / BM, 256, 0, stream>>>(Af, Wc, bias, hout);
        hin = hout;
    }

    // pooling + padded scatter + site gather
    zero_f32<<<2048, 256, 0, stream>>>(out_f, (size_t)out_size);
    zero_f32<<<1024, 256, 0, stream>>>(rsum, (size_t)NUM_RES * D + NUM_RES);
    atom_scatter<<<NUM_NODE / 8, 256, 0, stream>>>(hin, atom2residue, rsum, rcnt);
    graph_starts<<<1, 256, 0, stream>>>(residue2graph, starts);
    write_padded<<<NUM_RES, 256, 0, stream>>>(rsum, rcnt, residue2graph, starts, out_padded);
    site_gather<<<NUM_GRAPH * N_SITES, 256, 0, stream>>>(out_padded, positions, out_gf);
}
